// AECModel_5437428596950
// MI455X (gfx1250) — matrix operand order repaired and checked by reading, not compiled, not run on hardware
//
#include <hip/hip_runtime.h>
#include <hip/hip_bf16.h>

typedef _Float16 half_t;
typedef __attribute__((ext_vector_type(16))) _Float16 v16h;
typedef __attribute__((ext_vector_type(8)))  _Float16 v8h;
typedef __attribute__((ext_vector_type(8)))  float    v8f;

#define B_   2048
#define NF_  256
#define NT_  8
#define C_   32
#define T_   64
#define EPS_ 1e-5f

static __device__ __forceinline__ v8f wmma_f16(v16h a, v16h b, v8f c) {
    return __builtin_amdgcn_wmma_f32_16x16x32_f16(false, a, false, b, (short)0, c, false, false);
}

// Branch-free transcendentals: v_exp_f32 + v_rcp_f32 (TRANS32, co-executes
// with the WMMA pipe) instead of IEEE div sequence / tanhf branches.
static __device__ __forceinline__ float fast_sigmoid(float x) {
    return __builtin_amdgcn_rcpf(1.f + __expf(-x));
}
static __device__ __forceinline__ float fast_tanh(float x) {
    return 2.f * __builtin_amdgcn_rcpf(1.f + __expf(-2.f * x)) - 1.f;
}

// Load A fragment (16x32 f16) from row-major [rows][rowstride] memory, rows=batch.
// Layout: lanes 0-15 -> M=lane, K=0..15;
//         lanes 16-31 -> M=lane-16, K=16..31.
static __device__ __forceinline__ v16h load_a(const half_t* __restrict__ base, int rowstride, int lane) {
    const half_t* p = base + (lane & 15) * rowstride + ((lane < 16) ? 0 : 16);
    v8h lo = *(const v8h*)p;
    v8h hi = *(const v8h*)(p + 8);
    v16h a;
#pragma unroll
    for (int i = 0; i < 8; ++i) { a[i] = lo[i]; a[8 + i] = hi[i]; }
    return a;
}

// Store one 16x32 output tile from two D fragments (N-tiles 0/1) with optional
// ReLU and folded BatchNorm affine. out row stride = 32 f16.
static __device__ __forceinline__ void store_tile(half_t* __restrict__ po, v8f acc0, v8f acc1,
                                                  int lane, bool relu, bool post,
                                                  float sc0, float sh0, float sc1, float sh1) {
    const int n0 = lane & 15;
    const int mo = (lane < 16) ? 0 : 8;
#pragma unroll
    for (int r = 0; r < 8; ++r) {
        float x0 = acc0[r], x1 = acc1[r];
        if (relu) { x0 = fmaxf(x0, 0.f); x1 = fmaxf(x1, 0.f); }
        if (post) { x0 = x0 * sc0 + sh0; x1 = x1 * sc1 + sh1; }
        half_t* pr = po + (long long)(r + mo) * 32;
        pr[n0]      = (half_t)x0;
        pr[16 + n0] = (half_t)x1;
    }
}

// ---------------------------------------------------------------------------
// Frontend expand: BN over freq + Conv2d(1->32, kernel (1,8)) + ReLU.
// out layout: [f][b][32] f16
// ---------------------------------------------------------------------------
__global__ void __launch_bounds__(256)
expand_kernel(const float* __restrict__ x,
              const float* __restrict__ g, const float* __restrict__ be,
              const float* __restrict__ mn, const float* __restrict__ vr,
              const float* __restrict__ ew, const float* __restrict__ eb,
              half_t* __restrict__ out)
{
    int idx = blockIdx.x * blockDim.x + threadIdx.x;   // over B*NF
    int b = idx >> 8;
    int f = idx & 255;
    float sc = g[f] * rsqrtf(vr[f] + EPS_);
    float sh = be[f] - mn[f] * sc;
    float xt[NT_];
    const float* px = x + ((long long)b * NF_ + f) * NT_;
#pragma unroll
    for (int t = 0; t < NT_; ++t) xt[t] = px[t] * sc + sh;
    half_t o[C_];
#pragma unroll
    for (int c = 0; c < C_; ++c) {
        float a = eb[c];
#pragma unroll
        for (int t = 0; t < NT_; ++t) a += xt[t] * ew[c * NT_ + t];
        o[c] = (half_t)fmaxf(a, 0.f);
    }
    half_t* po = out + ((long long)f * B_ + b) * C_;
#pragma unroll
    for (int i = 0; i < 4; ++i) ((v8h*)po)[i] = ((v8h*)o)[i];
}

// ---------------------------------------------------------------------------
// Generic conv-as-GEMM over the batch dimension.
//   in : [Fin][B][KCH*32] f16      out: [Fout][B][32] f16
//   weight element (co, ci, tap d) at  w[co*wsO + ci*wsI + d*wsK + wofs]
//   normal : one wave = FPW output f positions x one 16-batch tile;
//            B fragments loaded once per wave. f_in = f*stride + d - pad
//            (zero-pad -> skip tap). grid = (Fout/FPW, 32).
//   CONVT  : stride-2 ConvTranspose(2,1). One wave = input position fin =
//            blockIdx.x; A loaded once, both output parities (f = 2*fin+p,
//            weight tap p) emitted. grid = (Fin, 32).
//   POSTAFF: fold eval-BatchNorm (per output channel) after ReLU.
// Block = 4 waves (one 16-batch tile each).
// ---------------------------------------------------------------------------
template<int TAPS, int KCH, bool CONVT, bool RELU, bool POSTAFF, int FPW>
__global__ void __launch_bounds__(128)
conv_gemm(const half_t* __restrict__ in, half_t* __restrict__ out,
          const float* __restrict__ w, const float* __restrict__ bias,
          int Fin, int stride, int pad,
          int wsO, int wsI, int wsK, int wofs,
          const float* __restrict__ bg, const float* __restrict__ bb,
          const float* __restrict__ bm, const float* __restrict__ bv)
{
    const int lane  = threadIdx.x & 31;
    const int wid   = threadIdx.x >> 5;
    const int btile = blockIdx.y * 4 + wid;
    const int Cin   = KCH * 32;
    const int n0    = lane & 15;
    const int kb    = (lane < 16) ? 0 : 16;
    const long long fstr = (long long)B_ * Cin;

    // B fragments: for CONVT load both parities; else one per tap.
    const int NTAP = CONVT ? 2 : TAPS;
    v16h bf[CONVT ? 2 : TAPS][KCH][2];
#pragma unroll
    for (int d = 0; d < NTAP; ++d)
#pragma unroll
        for (int kc = 0; kc < KCH; ++kc)
#pragma unroll
            for (int nt = 0; nt < 2; ++nt) {
                int n = nt * 16 + n0;
                v16h tmp;
#pragma unroll
                for (int k = 0; k < 16; ++k) {
                    int ci = kc * 32 + kb + k;
                    tmp[k] = (half_t)w[n * wsO + ci * wsI + d * wsK + wofs];
                }
                bf[d][kc][nt] = tmp;
            }

    const float bi0 = bias[n0], bi1 = bias[16 + n0];
    float sc0 = 1.f, sh0 = 0.f, sc1 = 1.f, sh1 = 0.f;
    if (POSTAFF) {
        sc0 = bg[n0] * rsqrtf(bv[n0] + EPS_);           sh0 = bb[n0] - bm[n0] * sc0;
        sc1 = bg[16 + n0] * rsqrtf(bv[16 + n0] + EPS_); sh1 = bb[16 + n0] - bm[16 + n0] * sc1;
    }

    if (CONVT) {
        const int fin = blockIdx.x;
        const half_t* pf = in + (long long)fin * fstr + (long long)(btile * 16) * Cin;
        v16h a[KCH];
#pragma unroll
        for (int kc = 0; kc < KCH; ++kc) a[kc] = load_a(pf + kc * 32, Cin, lane);
#pragma unroll
        for (int p = 0; p < 2; ++p) {
            v8f acc0, acc1;
#pragma unroll
            for (int r = 0; r < 8; ++r) { acc0[r] = bi0; acc1[r] = bi1; }
#pragma unroll
            for (int kc = 0; kc < KCH; ++kc) {
                acc0 = wmma_f16(a[kc], bf[p][kc][0], acc0);
                acc1 = wmma_f16(a[kc], bf[p][kc][1], acc1);
            }
            half_t* po = out + (long long)(2 * fin + p) * B_ * 32 + (long long)(btile * 16) * 32;
            store_tile(po, acc0, acc1, lane, RELU, POSTAFF, sc0, sh0, sc1, sh1);
        }
    } else {
#pragma unroll
        for (int ff = 0; ff < FPW; ++ff) {
            const int f = blockIdx.x * FPW + ff;
            v8f acc0, acc1;
#pragma unroll
            for (int r = 0; r < 8; ++r) { acc0[r] = bi0; acc1[r] = bi1; }
#pragma unroll
            for (int d = 0; d < TAPS; ++d) {
                int fi = f * stride + d - pad;
                if (fi < 0 || fi >= Fin) continue;
                const half_t* pf = in + (long long)fi * fstr + (long long)(btile * 16) * Cin;
#pragma unroll
                for (int kc = 0; kc < KCH; ++kc) {
                    v16h a = load_a(pf + kc * 32, Cin, lane);
                    acc0 = wmma_f16(a, bf[d][kc][0], acc0);
                    acc1 = wmma_f16(a, bf[d][kc][1], acc1);
                }
            }
            half_t* po = out + (long long)f * B_ * 32 + (long long)(btile * 16) * 32;
            store_tile(po, acc0, acc1, lane, RELU, POSTAFF, sc0, sh0, sc1, sh1);
        }
    }
}

// ---------------------------------------------------------------------------
// GRU core (torch gate order r|z|n). One wave = one 16-batch tile; hidden
// state lives in D-fragment registers; fed back as an A fragment through a
// per-wave LDS double buffer. 12 WMMAs (xi + gh, N=96) per step.
// in : [T][B][32] f16    out: [T][B][ostride] f16 at column offset oofs
// ---------------------------------------------------------------------------
template<bool POSTAFF>
static __device__ __forceinline__ void
gru_core(const half_t* __restrict__ in, half_t* __restrict__ out,
         int ostride, int oofs, int rev, int btile, int wid, int lane,
         half_t (*hbuf)[4][16][32],
         const float* __restrict__ wi, const float* __restrict__ wh,
         const float* __restrict__ bi, const float* __restrict__ bh,
         const float* __restrict__ bg, const float* __restrict__ bb,
         const float* __restrict__ bm, const float* __restrict__ bv)
{
    const int n0 = lane & 15;
    const int kb = (lane < 16) ? 0 : 16;
    const int mo = (lane < 16) ? 0 : 8;

    v16h wiF[6], whF[6];
#pragma unroll
    for (int nt = 0; nt < 6; ++nt) {
        int n = nt * 16 + n0;
        v16h t1, t2;
#pragma unroll
        for (int k = 0; k < 16; ++k) {
            t1[k] = (half_t)wi[n * 32 + kb + k];
            t2[k] = (half_t)wh[n * 32 + kb + k];
        }
        wiF[nt] = t1; whF[nt] = t2;
    }
    float biv[6], bhv[6];
#pragma unroll
    for (int nt = 0; nt < 6; ++nt) { biv[nt] = bi[nt * 16 + n0]; bhv[nt] = bh[nt * 16 + n0]; }

    float sc[2] = {1.f, 1.f}, sh[2] = {0.f, 0.f};
    if (POSTAFF) {
#pragma unroll
        for (int p = 0; p < 2; ++p) {
            int j = oofs + p * 16 + n0;
            sc[p] = bg[j] * rsqrtf(bv[j] + EPS_);
            sh[p] = bb[j] - bm[j] * sc[p];
        }
    }

    v8f hs[2];
#pragma unroll
    for (int r = 0; r < 8; ++r) { hs[0][r] = 0.f; hs[1][r] = 0.f; }
    const half_t hz = (half_t)0.f;
#pragma unroll
    for (int r = 0; r < 8; ++r) {
        hbuf[0][wid][r + mo][n0]      = hz;
        hbuf[0][wid][r + mo][16 + n0] = hz;
    }
    __syncthreads();

    for (int step = 0; step < T_; ++step) {
        int t  = rev ? (T_ - 1 - step) : step;
        int pb = step & 1;

        const half_t* pa = in + ((long long)t * B_ + btile * 16) * 32;
        v16h ax = load_a(pa, 32, lane);
        v16h ah = load_a(&hbuf[pb][wid][0][0], 32, lane);

        v8f hn[2];
#pragma unroll
        for (int p = 0; p < 2; ++p) {
            v8f rx, zx, nx, rh, zh, nh;
#pragma unroll
            for (int r = 0; r < 8; ++r) {
                rx[r] = biv[p];     zx[r] = biv[2 + p]; nx[r] = biv[4 + p];
                rh[r] = bhv[p];     zh[r] = bhv[2 + p]; nh[r] = bhv[4 + p];
            }
            rx = wmma_f16(ax, wiF[p],     rx);  rh = wmma_f16(ah, whF[p],     rh);
            zx = wmma_f16(ax, wiF[2 + p], zx);  zh = wmma_f16(ah, whF[2 + p], zh);
            nx = wmma_f16(ax, wiF[4 + p], nx);  nh = wmma_f16(ah, whF[4 + p], nh);
#pragma unroll
            for (int r = 0; r < 8; ++r) {
                float rg = fast_sigmoid(rx[r] + rh[r]);
                float zg = fast_sigmoid(zx[r] + zh[r]);
                float ng = fast_tanh(nx[r] + rg * nh[r]);
                hn[p][r] = (1.f - zg) * ng + zg * hs[p][r];
            }
        }
        hs[0] = hn[0]; hs[1] = hn[1];

#pragma unroll
        for (int r = 0; r < 8; ++r) {
            hbuf[1 - pb][wid][r + mo][n0]      = (half_t)hn[0][r];
            hbuf[1 - pb][wid][r + mo][16 + n0] = (half_t)hn[1][r];
        }
        half_t* po = out + ((long long)t * B_ + btile * 16) * ostride + oofs;
#pragma unroll
        for (int r = 0; r < 8; ++r) {
            float y0 = hn[0][r], y1 = hn[1][r];
            if (POSTAFF) { y0 = y0 * sc[0] + sh[0]; y1 = y1 * sc[1] + sh[1]; }
            half_t* pr = po + (long long)(r + mo) * ostride;
            pr[n0]      = (half_t)y0;
            pr[16 + n0] = (half_t)y1;
        }
        __syncthreads();
    }
}

// gru_u: unidirectional, no post-affine. grid = 32 blocks of 4 waves.
__global__ void __launch_bounds__(128)
gru_uni(const half_t* __restrict__ in, half_t* __restrict__ out,
        const float* __restrict__ wi, const float* __restrict__ wh,
        const float* __restrict__ bi, const float* __restrict__ bh)
{
    __shared__ half_t hbuf[2][4][16][32];
    const int lane = threadIdx.x & 31;
    const int wid  = threadIdx.x >> 5;
    const int btile = blockIdx.x * 4 + wid;
    gru_core<false>(in, out, 32, 0, 0, btile, wid, lane, hbuf,
                    wi, wh, bi, bh, nullptr, nullptr, nullptr, nullptr);
}

// Fused bidirectional pass: blocks 0..31 forward (features 0..31),
// blocks 32..63 reverse (features 32..63); bn_post folded into stores.
__global__ void __launch_bounds__(128)
gru_bidir(const half_t* __restrict__ in, half_t* __restrict__ out,
          const float* __restrict__ wi_f, const float* __restrict__ wh_f,
          const float* __restrict__ bi_f, const float* __restrict__ bh_f,
          const float* __restrict__ wi_r, const float* __restrict__ wh_r,
          const float* __restrict__ bi_r, const float* __restrict__ bh_r,
          const float* __restrict__ bg, const float* __restrict__ bb,
          const float* __restrict__ bm, const float* __restrict__ bv)
{
    __shared__ half_t hbuf[2][4][16][32];
    const int lane  = threadIdx.x & 31;
    const int wid   = threadIdx.x >> 5;
    const int rev   = (blockIdx.x >= 32) ? 1 : 0;
    const int btile = (blockIdx.x & 31) * 4 + wid;
    const float* wi = rev ? wi_r : wi_f;
    const float* wh = rev ? wh_r : wh_f;
    const float* bi = rev ? bi_r : bi_f;
    const float* bh = rev ? bh_r : bh_f;
    gru_core<true>(in, out, 64, rev ? 32 : 0, rev, btile, wid, lane, hbuf,
                   wi, wh, bi, bh, bg, bb, bm, bv);
}

// ---------------------------------------------------------------------------
// Head: 1x1 conv 32->2 + sigmoid, write fp32 [B,2,256,1].
// ---------------------------------------------------------------------------
__global__ void __launch_bounds__(256)
out_kernel(const half_t* __restrict__ in,
           const float* __restrict__ w, const float* __restrict__ bias,
           float* __restrict__ out)
{
    int idx = blockIdx.x * blockDim.x + threadIdx.x;   // over B*NF
    int b = idx >> 8;
    int f = idx & 255;
    const half_t* p = in + ((long long)f * B_ + b) * 32;
    float a0 = bias[0], a1 = bias[1];
#pragma unroll
    for (int c = 0; c < 32; ++c) {
        float x = (float)p[c];
        a0 += x * w[c];
        a1 += x * w[32 + c];
    }
    out[(long long)b * 512 + f]       = fast_sigmoid(a0);
    out[(long long)b * 512 + 256 + f] = fast_sigmoid(a1);
}

// ---------------------------------------------------------------------------
extern "C" void kernel_launch(void* const* d_in, const int* in_sizes, int n_in,
                              void* d_out, int out_size, void* d_ws, size_t ws_size,
                              hipStream_t stream) {
    (void)in_sizes; (void)n_in; (void)out_size; (void)ws_size;

    const float* x        = (const float*)d_in[0];
    const float* bn_in_g  = (const float*)d_in[1];
    const float* bn_in_b  = (const float*)d_in[2];
    const float* bn_in_m  = (const float*)d_in[3];
    const float* bn_in_v  = (const float*)d_in[4];
    const float* expand_w = (const float*)d_in[5];
    const float* expand_b = (const float*)d_in[6];
    const float* conv1_w  = (const float*)d_in[7];
    const float* conv1_b  = (const float*)d_in[8];
    const float* conv2_w  = (const float*)d_in[9];
    const float* conv2_b  = (const float*)d_in[10];
    const float* down1_w  = (const float*)d_in[11];
    const float* down1_b  = (const float*)d_in[12];
    const float* conv3_w  = (const float*)d_in[13];
    const float* conv3_b  = (const float*)d_in[14];
    const float* down2_w  = (const float*)d_in[15];
    const float* down2_b  = (const float*)d_in[16];
    const float* conv4_w  = (const float*)d_in[17];
    const float* conv4_b  = (const float*)d_in[18];
    const float* bn_pre_g = (const float*)d_in[19];
    const float* bn_pre_b = (const float*)d_in[20];
    const float* bn_pre_m = (const float*)d_in[21];
    const float* bn_pre_v = (const float*)d_in[22];
    const float* gru_u_wi = (const float*)d_in[23];
    const float* gru_u_wh = (const float*)d_in[24];
    const float* gru_u_bi = (const float*)d_in[25];
    const float* gru_u_bh = (const float*)d_in[26];
    const float* gru_f_wi = (const float*)d_in[27];
    const float* gru_f_wh = (const float*)d_in[28];
    const float* gru_f_bi = (const float*)d_in[29];
    const float* gru_f_bh = (const float*)d_in[30];
    const float* gru_r_wi = (const float*)d_in[31];
    const float* gru_r_wh = (const float*)d_in[32];
    const float* gru_r_bi = (const float*)d_in[33];
    const float* gru_r_bh = (const float*)d_in[34];
    const float* bn_po_g  = (const float*)d_in[35];
    const float* bn_po_b  = (const float*)d_in[36];
    const float* bn_po_m  = (const float*)d_in[37];
    const float* bn_po_v  = (const float*)d_in[38];
    const float* up1_w    = (const float*)d_in[39];
    const float* up1_b    = (const float*)d_in[40];
    const float* conv5_w  = (const float*)d_in[41];
    const float* conv5_b  = (const float*)d_in[42];
    const float* up2_w    = (const float*)d_in[43];
    const float* up2_b    = (const float*)d_in[44];
    const float* conv6_w  = (const float*)d_in[45];
    const float* conv6_b  = (const float*)d_in[46];
    const float* out_w    = (const float*)d_in[47];
    const float* out_b    = (const float*)d_in[48];
    float* y = (float*)d_out;

    // Workspace (67 MB total, carefully overlapped):
    half_t* bufA  = (half_t*)d_ws;                       // [256][B][32] f16
    half_t* bufB  = bufA + (size_t)NF_ * B_ * 32;        // [256][B][32] f16
    half_t* bufS  = bufA;                                 // [64][B][32]  (encoder out; A dead)
    half_t* bufU  = bufS + (size_t)T_ * B_ * 32;         // [64][B][32]  (gru_u out)
    half_t* bufS2 = bufU + (size_t)T_ * B_ * 32;         // [64][B][64]  (bi-GRU + bn_post)
    half_t* bufD  = bufA;                                 // decoder mid [128][B][32]

    dim3 blk128(128);

    // Frontend
    expand_kernel<<<(B_ * NF_) / 256, 256, 0, stream>>>(
        x, bn_in_g, bn_in_b, bn_in_m, bn_in_v, expand_w, expand_b, bufA);

    conv_gemm<1,1,false,true,false,4><<<dim3(64,32), blk128, 0, stream>>>(
        bufA, bufB, conv1_w, conv1_b, 256, 1, 0, 32, 1, 0, 0, nullptr, nullptr, nullptr, nullptr);
    conv_gemm<3,1,false,true,false,4><<<dim3(64,32), blk128, 0, stream>>>(
        bufB, bufA, conv2_w, conv2_b, 256, 1, 1, 96, 3, 1, 0, nullptr, nullptr, nullptr, nullptr);
    conv_gemm<2,1,false,true,false,4><<<dim3(32,32), blk128, 0, stream>>>(
        bufA, bufB, down1_w, down1_b, 256, 2, 0, 64, 2, 1, 0, nullptr, nullptr, nullptr, nullptr);
    conv_gemm<3,1,false,true,false,4><<<dim3(32,32), blk128, 0, stream>>>(
        bufB, bufA, conv3_w, conv3_b, 128, 1, 1, 96, 3, 1, 0, nullptr, nullptr, nullptr, nullptr);
    conv_gemm<2,1,false,true,false,4><<<dim3(16,32), blk128, 0, stream>>>(
        bufA, bufB, down2_w, down2_b, 128, 2, 0, 64, 2, 1, 0, nullptr, nullptr, nullptr, nullptr);
    // conv4 + ReLU + bn_pre folded
    conv_gemm<3,1,false,true,true,4><<<dim3(16,32), blk128, 0, stream>>>(
        bufB, bufS, conv4_w, conv4_b, 64, 1, 1, 96, 3, 1, 0, bn_pre_g, bn_pre_b, bn_pre_m, bn_pre_v);

    // GRUs: u, then fused forward/reverse with bn_post folded
    gru_uni<<<32, blk128, 0, stream>>>(bufS, bufU, gru_u_wi, gru_u_wh, gru_u_bi, gru_u_bh);
    gru_bidir<<<64, blk128, 0, stream>>>(bufU, bufS2,
        gru_f_wi, gru_f_wh, gru_f_bi, gru_f_bh,
        gru_r_wi, gru_r_wh, gru_r_bi, gru_r_bh,
        bn_po_g, bn_po_b, bn_po_m, bn_po_v);

    // Decoder: up1 (ConvT 64->32, both parities per wave), conv5 (3x3, kw=1 live),
    // up2 (ConvT 32->32), conv6
    conv_gemm<1,2,true,true,false,1><<<dim3(64,32), blk128, 0, stream>>>(
        bufS2, bufB, up1_w, up1_b, 64, 0, 0, 2, 64, 1, 0, nullptr, nullptr, nullptr, nullptr);
    conv_gemm<3,1,false,true,false,4><<<dim3(32,32), blk128, 0, stream>>>(
        bufB, bufD, conv5_w, conv5_b, 128, 1, 1, 288, 9, 3, 1, nullptr, nullptr, nullptr, nullptr);
    conv_gemm<1,1,true,true,false,1><<<dim3(128,32), blk128, 0, stream>>>(
        bufD, bufB, up2_w, up2_b, 128, 0, 0, 2, 64, 1, 0, nullptr, nullptr, nullptr, nullptr);
    conv_gemm<3,1,false,true,false,4><<<dim3(64,32), blk128, 0, stream>>>(
        bufB, bufA, conv6_w, conv6_b, 256, 1, 1, 288, 9, 3, 1, nullptr, nullptr, nullptr, nullptr);

    // Head
    out_kernel<<<(B_ * NF_) / 256, 256, 0, stream>>>(bufA, out_w, out_b, y);
}
